// DecoderLayer_4011499455210
// MI455X (gfx1250) — compile-verified, matrix operand order checked by reading, not run on hardware
//
#include <hip/hip_runtime.h>

// ---------------------------------------------------------------------------
// DecoderLayer on MI455X (gfx1250): bf16 WMMA GEMMs + 1-wave flash attention.
// B=2, S=2048, DIM=1024, H=16, DH=64, DFF=2048.
// Matrix math: v_wmma_f32_16x16x32_bf16 (wave32, f32 accumulate).
// Tile staging: GLOBAL_LOAD_ASYNC_TO_LDS_B128 (ASYNCcnt) + double-buffered LDS.
// ---------------------------------------------------------------------------

typedef __attribute__((ext_vector_type(16))) __bf16 v16bf;
typedef __attribute__((ext_vector_type(8)))  float  v8f;
typedef int v4i_vec __attribute__((vector_size(16)));

union FragU { v16bf v; uint4 u[2]; };

__device__ __forceinline__ unsigned short f32_to_bf16(float f) {
  unsigned int u = __float_as_uint(f);
  u += 0x7FFFu + ((u >> 16) & 1u);            // round-to-nearest-even
  return (unsigned short)(u >> 16);
}

__device__ __forceinline__ v8f zero8() {
  v8f z = {0.f, 0.f, 0.f, 0.f, 0.f, 0.f, 0.f, 0.f};
  return z;
}

// 16-byte global -> LDS copy. Prefer the gfx1250 async path (ASYNCcnt),
// fall back to a synchronous VGPR round-trip if the builtin is absent.
// Builtin signature (from compiler diagnostic): param 0 is a pointer to
// int __attribute__((vector_size(16))) in the global address space.
__device__ __forceinline__ void copy16_g2l(unsigned short* ldst,
                                           const unsigned short* gsrc) {
#if __has_builtin(__builtin_amdgcn_global_load_async_to_lds_b128)
  __builtin_amdgcn_global_load_async_to_lds_b128(
      (__attribute__((address_space(1))) v4i_vec*)gsrc,
      (__attribute__((address_space(3))) v4i_vec*)ldst, 0, 0);
#else
  *(uint4*)ldst = *(const uint4*)gsrc;
#endif
}

__device__ __forceinline__ void wait_async0() {
#if __has_builtin(__builtin_amdgcn_s_wait_asynccnt)
  __builtin_amdgcn_s_wait_asynccnt(0);
#elif __has_builtin(__builtin_amdgcn_global_load_async_to_lds_b128)
  asm volatile("s_wait_asynccnt 0x0" ::: "memory");
#endif
}

// Load one 16x32 bf16 WMMA operand fragment from a row-major [rows x ld]
// bf16 buffer (global or LDS). ISA layout (CDNA5 ISA):
//   lane L: row = L&15 ; K-halves selected by L>>4 ; pairs packed per VGPR.
// Per lane this is two contiguous 16-byte loads (K = k0..k0+7, k0+16..k0+23).
__device__ __forceinline__ v16bf load_frag(const unsigned short* p, int ld,
                                           int row, int kBase, int lane) {
  int r  = row + (lane & 15);
  int k0 = kBase + ((lane >> 4) << 3);
  FragU f;
  f.u[0] = *(const uint4*)(p + (size_t)r * ld + k0);
  f.u[1] = *(const uint4*)(p + (size_t)r * ld + k0 + 16);
  return f.v;
}

__device__ __forceinline__ v8f wmma_bf16(v16bf a, v16bf b, v8f c) {
  // (neg_a, A, neg_b, B, c_mod, C, reuse_a, reuse_b)
  return __builtin_amdgcn_wmma_f32_16x16x32_bf16(false, a, false, b,
                                                 (short)0, c, false, false);
}

// ---------------------------------------------------------------------------
// Generic bf16 GEMM core: C[M,N] = A[M,K] * B[K,N] (+bias, +relu).
// Block = 256 threads = 8 waves; wave tile WTM x WTN (MI x NI accumulators).
// A tile staged row-major via async global->LDS copies; B tile staged
// transposed (N-major) so B fragments load exactly like A fragments.
// Double-buffered LDS: next K-step's copies issued before current compute.
// ---------------------------------------------------------------------------
template <int BM, int BN, int WTM, int WTN>
__device__ __forceinline__ void gemm_core(
    const unsigned short* __restrict__ A, int lda,
    const unsigned short* __restrict__ B, int ldb,
    const float* __restrict__ bias, bool relu,
    float* __restrict__ Cf, unsigned short* __restrict__ Cb, int ldc,
    int K, int tileM, int tileN,
    unsigned short* lA, unsigned short* lBt) {
  constexpr int MI = WTM / 16, NI = WTN / 16;
  constexpr int WavesN = BN / WTN;
  constexpr int ABUF = BM * 40;     // shorts per A buffer (stride 40 = pad)
  constexpr int BBUF = BN * 40;
  static_assert((BM / WTM) * (BN / WTN) == 8, "8 waves per block");

  const int tid  = threadIdx.x;
  const int lane = tid & 31;
  const int wave = tid >> 5;
  const int wm = (wave / WavesN) * WTM;
  const int wn = (wave % WavesN) * WTN;
  const int rowBase = tileM * BM;
  const int colBase = tileN * BN;

  v8f acc[MI][NI];
#pragma unroll
  for (int i = 0; i < MI; ++i)
#pragma unroll
    for (int j = 0; j < NI; ++j) acc[i][j] = zero8();

  auto stage = [&](int kb, int buf) {
    unsigned short* dA = lA + buf * ABUF;
    unsigned short* dB = lBt + buf * BBUF;
    // A tile (BM x 32): contiguous 16B rows-of-K chunks, async to LDS.
    for (int idx = tid * 8; idx < BM * 32; idx += 256 * 8) {
      int r = idx >> 5, k = idx & 31;
      copy16_g2l(dA + r * 40 + k,
                 A + (size_t)(rowBase + r) * lda + kb + k);
    }
    // B tile (32 x BN) transposed into dB[BN][40] (N-major).
    for (int idx = tid * 8; idx < 32 * BN; idx += 256 * 8) {
      int k = idx / BN, n = idx % BN;
      unsigned short tmp[8];
      *(uint4*)tmp = *(const uint4*)(B + (size_t)(kb + k) * ldb + colBase + n);
#pragma unroll
      for (int j = 0; j < 8; ++j) dB[(n + j) * 40 + k] = tmp[j];
    }
  };

  stage(0, 0);
  int buf = 0;
  for (int kb = 0; kb < K; kb += 32, buf ^= 1) {
    wait_async0();        // own async copies for `buf` complete
    __syncthreads();      // all waves' stores to `buf` visible
    if (kb + 32 < K) stage(kb + 32, buf ^ 1);   // prefetch next K-step

    const unsigned short* sA = lA + buf * ABUF;
    const unsigned short* sB = lBt + buf * BBUF;
    v16bf af[MI], bfr[NI];
#pragma unroll
    for (int i = 0; i < MI; ++i) af[i] = load_frag(sA, 40, wm + i * 16, 0, lane);
#pragma unroll
    for (int j = 0; j < NI; ++j) bfr[j] = load_frag(sB, 40, wn + j * 16, 0, lane);
#pragma unroll
    for (int i = 0; i < MI; ++i)
#pragma unroll
      for (int j = 0; j < NI; ++j)
        acc[i][j] = wmma_bf16(af[i], bfr[j], acc[i][j]);
  }

  // Epilogue. C fragment layout: elem e -> row = (lane>>4)*8+e, col = lane&15.
#pragma unroll
  for (int i = 0; i < MI; ++i)
#pragma unroll
    for (int j = 0; j < NI; ++j)
#pragma unroll
      for (int e = 0; e < 8; ++e) {
        int row = rowBase + wm + i * 16 + ((lane >> 4) << 3) + e;
        int col = colBase + wn + j * 16 + (lane & 15);
        float v = acc[i][j][e];
        if (bias) v += bias[col];
        if (relu) v = fmaxf(v, 0.0f);
        if (Cf) Cf[(size_t)row * ldc + col] = v;
        if (Cb) Cb[(size_t)row * ldc + col] = f32_to_bf16(v);
      }
}

// FFN GEMMs: 128x128 block, 64x32 wave tiles (8 WMMA / 6 frag loads per step).
__global__ __launch_bounds__(256) void gemm_bf16_kernel(
    const unsigned short* __restrict__ A, int lda,
    const unsigned short* __restrict__ B, int ldb,
    const float* __restrict__ bias, int relu,
    float* __restrict__ Cf, unsigned short* __restrict__ Cb, int ldc, int K) {
  __shared__ __align__(16) unsigned short lA[2 * 128 * 40];
  __shared__ __align__(16) unsigned short lBt[2 * 128 * 40];
  gemm_core<128, 128, 64, 32>(A, lda, B, ldb, bias, relu != 0, Cf, Cb, ldc, K,
                              blockIdx.x, blockIdx.y, lA, lBt);
}

// Per-head QKV projection: out[b,h,s,:] = X[b,s,:] @ W[h,:,:]
// X: bf16 [B*S,1024], W: bf16 [H,1024,64], out: bf16 [B,H,S,64].
__global__ __launch_bounds__(256) void proj_kernel(
    const unsigned short* __restrict__ X, const unsigned short* __restrict__ W,
    unsigned short* __restrict__ Out) {
  __shared__ __align__(16) unsigned short lA[2 * 128 * 40];
  __shared__ __align__(16) unsigned short lBt[2 * 64 * 40];
  const int b = blockIdx.y, h = blockIdx.z;
  const unsigned short* Ab = X + (size_t)b * 2048 * 1024;
  const unsigned short* Bb = W + (size_t)h * 1024 * 64;
  unsigned short* Cb = Out + (size_t)(b * 16 + h) * 2048 * 64;
  gemm_core<128, 64, 32, 32>(Ab, 1024, Bb, 64, nullptr, false, nullptr, Cb, 64,
                             1024, blockIdx.x, 0, lA, lBt);
}

// ---------------------------------------------------------------------------
// Flash attention, 1 wave (32 threads) per 16-row Q tile per (b,h).
// Q,K,V: bf16 [B*H, S, 64]; Out: bf16 [B, S, H*64] (heads concatenated).
// Keys processed 32 at a time; online softmax stats kept in the native
// C-fragment lane layout (8 rows per lane), reduced with shfl_xor width-16.
// ---------------------------------------------------------------------------
__global__ __launch_bounds__(32) void attn_kernel(
    const unsigned short* __restrict__ Q, const unsigned short* __restrict__ Km,
    const unsigned short* __restrict__ Vm, unsigned short* __restrict__ Out,
    int S, int causal) {
  __shared__ __align__(16) unsigned short sP[16 * 40];   // P tile 16x32 (bf16)
  __shared__ __align__(16) unsigned short sVt[64 * 40];  // V^T tile 64x32

  const int lane = threadIdx.x & 31;
  const int bh = blockIdx.y;
  const int b = bh >> 4;         // H == 16
  const int h = bh & 15;
  const size_t base = (size_t)bh * S * 64;
  const int q0 = blockIdx.x * 16;

  const v16bf qa0 = load_frag(Q + base, 64, q0, 0, lane);
  const v16bf qa1 = load_frag(Q + base, 64, q0, 32, lane);

  v8f acc[4];
#pragma unroll
  for (int n = 0; n < 4; ++n) acc[n] = zero8();
  float m[8], l[8];
#pragma unroll
  for (int e = 0; e < 8; ++e) { m[e] = -3.0e38f; l[e] = 0.0f; }

  const int tmax = causal ? ((q0 + 15) >> 5) : (S / 32 - 1);
  for (int t = 0; t <= tmax; ++t) {
    const int key0 = t * 32;

    // Scores: 16 queries x 32 keys = two 16x16 WMMA tiles, K-dim = DH = 64.
    // K is [keys,64] row-major: per lane, column=key, dh contiguous — already
    // the transposed layout the B operand wants.
    v8f sc[2];
#pragma unroll
    for (int j = 0; j < 2; ++j) {
      v16bf kb0 = load_frag(Km + base, 64, key0 + j * 16, 0, lane);
      v16bf kb1 = load_frag(Km + base, 64, key0 + j * 16, 32, lane);
      v8f s = zero8();
      s = wmma_bf16(qa0, kb0, s);
      s = wmma_bf16(qa1, kb1, s);
      sc[j] = s;
    }

    // Scale + causal mask.
    float p[2][8];
#pragma unroll
    for (int e = 0; e < 8; ++e) {
      int row = q0 + ((lane >> 4) << 3) + e;
#pragma unroll
      for (int j = 0; j < 2; ++j) {
        int col = key0 + j * 16 + (lane & 15);
        float v = sc[j][e] * 0.125f;   // 1/sqrt(64)
        if (causal && col > row) v = -1.0e9f;
        p[j][e] = v;
      }
    }

    // Online softmax update + write P (bf16) into LDS in A-fragment order.
#pragma unroll
    for (int e = 0; e < 8; ++e) {
      float rm = fmaxf(p[0][e], p[1][e]);
#pragma unroll
      for (int msk = 8; msk >= 1; msk >>= 1)
        rm = fmaxf(rm, __shfl_xor(rm, msk, 32));
      float mnew = fmaxf(m[e], rm);
      float es = __expf(m[e] - mnew);
      float p0 = __expf(p[0][e] - mnew);
      float p1 = __expf(p[1][e] - mnew);
      float rs = p0 + p1;
#pragma unroll
      for (int msk = 8; msk >= 1; msk >>= 1)
        rs += __shfl_xor(rs, msk, 32);
      l[e] = l[e] * es + rs;
      m[e] = mnew;
#pragma unroll
      for (int n = 0; n < 4; ++n) acc[n][e] *= es;
      int prow = ((lane >> 4) << 3) + e;
      sP[prow * 40 + (lane & 15)]      = f32_to_bf16(p0);
      sP[prow * 40 + 16 + (lane & 15)] = f32_to_bf16(p1);
    }

    // Stage V tile transposed: sVt[dh][key] for the P@V B operand.
    for (int i = lane * 8; i < 32 * 64; i += 32 * 8) {
      int key = i >> 6, d0 = i & 63;
      unsigned short tmp[8];
      *(uint4*)tmp = *(const uint4*)(Vm + base + (size_t)(key0 + key) * 64 + d0);
#pragma unroll
      for (int j2 = 0; j2 < 8; ++j2) sVt[(d0 + j2) * 40 + key] = tmp[j2];
    }
    __syncthreads();

    // O += P(16x32) @ V(32x64): four 16x16 output tiles, one K=32 step.
    v16bf pa = load_frag(sP, 40, 0, 0, lane);
#pragma unroll
    for (int n = 0; n < 4; ++n) {
      v16bf vb = load_frag(sVt, 40, n * 16, 0, lane);
      acc[n] = wmma_bf16(pa, vb, acc[n]);
    }
    __syncthreads();
  }

  // Normalize and write concat-head output (bf16, feeds next GEMM).
#pragma unroll
  for (int e = 0; e < 8; ++e) {
    float inv = 1.0f / l[e];
    int row = q0 + ((lane >> 4) << 3) + e;
#pragma unroll
    for (int n = 0; n < 4; ++n) {
      int col = h * 64 + n * 16 + (lane & 15);
      Out[((size_t)b * S + row) * 1024 + col] = f32_to_bf16(acc[n][e] * inv);
    }
  }
}

// fp32 -> bf16 bulk convert (4 elements/thread).
__global__ __launch_bounds__(256) void cvt_kernel(const float* __restrict__ in,
                                                  unsigned short* __restrict__ out,
                                                  int n4) {
  int i = blockIdx.x * blockDim.x + threadIdx.x;
  if (i < n4) {
    float4 f = ((const float4*)in)[i];
    unsigned short r[4] = {f32_to_bf16(f.x), f32_to_bf16(f.y),
                           f32_to_bf16(f.z), f32_to_bf16(f.w)};
    ((uint2*)out)[i] = *(uint2*)r;
  }
}

// ---------------------------------------------------------------------------
extern "C" void kernel_launch(void* const* d_in, const int* in_sizes, int n_in,
                              void* d_out, int out_size, void* d_ws, size_t ws_size,
                              hipStream_t stream) {
  (void)in_sizes; (void)n_in; (void)out_size; (void)ws_size;
  constexpr int Bd = 2, S = 2048, DIM = 1024, H = 16, DH = 64, DFF = 2048;
  constexpr int M = Bd * S;  // 4096 rows

  const float* de_x = (const float*)d_in[0];
  const float* en_x = (const float*)d_in[1];
  // d_in[2] = mask: causal mask applied analytically in attn_kernel.
  const float* Wq = (const float*)d_in[3];
  const float* Wk = (const float*)d_in[4];
  const float* Wv = (const float*)d_in[5];
  const float* W1 = (const float*)d_in[6];
  const float* b1 = (const float*)d_in[7];
  const float* W2 = (const float*)d_in[8];
  const float* b2 = (const float*)d_in[9];
  float* y = (float*)d_out;

  char* ws = (char*)d_ws;
  size_t off = 0;
  auto alloc = [&](size_t elems) {
    unsigned short* p = (unsigned short*)(ws + off);
    off += ((elems * 2 + 255) & ~(size_t)255);
    return p;
  };
  unsigned short* xaB = alloc((size_t)M * DIM);        // de_x bf16
  unsigned short* xbB = alloc((size_t)M * DIM);        // en_x bf16
  unsigned short* wqB = alloc((size_t)H * DIM * DH);
  unsigned short* wkB = alloc((size_t)H * DIM * DH);
  unsigned short* wvB = alloc((size_t)H * DIM * DH);
  unsigned short* w1B = alloc((size_t)DIM * DFF);
  unsigned short* w2B = alloc((size_t)DFF * DIM);
  unsigned short* qB  = alloc((size_t)Bd * H * S * DH);
  unsigned short* kB  = alloc((size_t)Bd * H * S * DH);
  unsigned short* vB  = alloc((size_t)Bd * H * S * DH);
  unsigned short* hB  = alloc((size_t)M * DIM);        // h1 then h2
  unsigned short* ffB = alloc((size_t)M * DFF);        // relu(h2 W1 + b1)

  auto cvt = [&](const float* src, unsigned short* dst, int n) {
    int n4 = n / 4;
    cvt_kernel<<<dim3((n4 + 255) / 256), dim3(256), 0, stream>>>(src, dst, n4);
  };
  cvt(de_x, xaB, M * DIM);
  cvt(en_x, xbB, M * DIM);
  cvt(Wq, wqB, H * DIM * DH);
  cvt(Wk, wkB, H * DIM * DH);
  cvt(Wv, wvB, H * DIM * DH);
  cvt(W1, w1B, DIM * DFF);
  cvt(W2, w2B, DFF * DIM);

  const dim3 pg(S / 128, Bd, H);
  const dim3 ag(S / 16, Bd * H);

  // Pass 1: h1 = MHA(q=de, k=de, v=de, causal)
  proj_kernel<<<pg, 256, 0, stream>>>(xaB, wqB, qB);
  proj_kernel<<<pg, 256, 0, stream>>>(xaB, wkB, kB);
  proj_kernel<<<pg, 256, 0, stream>>>(xaB, wvB, vB);
  attn_kernel<<<ag, 32, 0, stream>>>(qB, kB, vB, hB, S, 1);

  // Pass 2: h2 = MHA(q=en, k=en, v=h1, no mask), same weights
  proj_kernel<<<pg, 256, 0, stream>>>(xbB, wqB, qB);
  proj_kernel<<<pg, 256, 0, stream>>>(xbB, wkB, kB);
  proj_kernel<<<pg, 256, 0, stream>>>(hB, wvB, vB);
  attn_kernel<<<ag, 32, 0, stream>>>(qB, kB, vB, hB, S, 0);

  // FFN: y = relu(h2 W1 + b1) W2 + b2
  gemm_bf16_kernel<<<dim3(M / 128, DFF / 128), 256, 0, stream>>>(
      hB, DIM, w1B, DFF, b1, 1, nullptr, ffB, DFF, DIM);
  gemm_bf16_kernel<<<dim3(M / 128, DIM / 128), 256, 0, stream>>>(
      ffB, DFF, w2B, DIM, b2, 0, y, nullptr, DIM, DFF);
}
